// CrossAttention_24292335026756
// MI455X (gfx1250) — matrix operand order repaired and checked by reading, not compiled, not run on hardware
//
#include <hip/hip_runtime.h>
#include <hip/hip_bf16.h>

typedef __attribute__((ext_vector_type(16))) _Float16 v16h;
typedef __attribute__((ext_vector_type(8)))  _Float16 v8h;
typedef __attribute__((ext_vector_type(8)))  float    v8f;
typedef __attribute__((ext_vector_type(4)))  float    v4f;
typedef __attribute__((ext_vector_type(4)))  int      v4i;

#define N_  4
#define VQ_ 4096
#define VK_ 4096
#define C_  128
#define H_  64

// ---- CDNA5 async global->LDS copy (ASYNCcnt path), guarded by builtin probe ----
#if defined(__has_builtin)
#if __has_builtin(__builtin_amdgcn_global_load_async_to_lds_b128)
#define ASYNC_LDS 1
#endif
#if __has_builtin(__builtin_amdgcn_s_wait_asynccnt)
#define HAVE_WAIT_ASYNC_BUILTIN 1
#endif
#endif

#ifdef ASYNC_LDS
template <int OFF>
__device__ __forceinline__ void async_cp16(const _Float16* g, _Float16* l) {
  __builtin_amdgcn_global_load_async_to_lds_b128(
      (v4i __attribute__((address_space(1)))*)g,
      (v4i __attribute__((address_space(3)))*)l, OFF, 0);
}
__device__ __forceinline__ void wait_async0() {
#ifdef HAVE_WAIT_ASYNC_BUILTIN
  __builtin_amdgcn_s_wait_asynccnt(0);
#else
  asm volatile("s_wait_asynccnt 0x0" ::: "memory");
#endif
}
#endif

union V16U { v16h v; v8h h[2]; };

__device__ __forceinline__ v8f wmma_f16(v16h a, v16h b, v8f c) {
  return __builtin_amdgcn_wmma_f32_16x16x32_f16(false, a, false, b, (short)0, c,
                                                false, false);
}

// A fragment (16x32 f16): lane holds row lane%16; g=0: K 0..7 & 16..23, g=1: K 8..15 & 24..31
__device__ __forceinline__ v16h load_a32(const _Float16* row, int g) {
  V16U u;
  u.h[0] = *(const v8h*)(row + g * 8);
  u.h[1] = *(const v8h*)(row + 16 + g * 8);
  return u.v;
}

// B fragment (32x16 f16): lane holds col lane%16; g=0: K 0..7 & 16..23, g=1: K 8..15 & 24..31
__device__ __forceinline__ v16h load_b32(const _Float16* row, int g) {
  V16U u;
  const _Float16* p = row + g * 8;
  u.h[0] = *(const v8h*)(p);
  u.h[1] = *(const v8h*)(p + 16);
  return u.v;
}

__device__ __forceinline__ v8h cvt8(const float* p) {
  v4f a = *(const v4f*)p;
  v4f b = *(const v4f*)(p + 4);
  v8h r;
#pragma unroll
  for (int i = 0; i < 4; ++i) { r[i] = (_Float16)a[i]; r[i + 4] = (_Float16)b[i]; }
  return r;
}

// hardware base-2 exponential (v_exp_f32); scores are pre-scaled by log2(e)
__device__ __forceinline__ float exp2_fast(float x) {
  return __builtin_amdgcn_exp2f(x);
}

// ---- DPP ROW_XMASK butterfly reductions over the 16-lane half-groups (pure VALU) ----
template <int CTRL>
__device__ __forceinline__ float dpp_mov(float v) {
  return __int_as_float(__builtin_amdgcn_update_dpp(
      0, __float_as_int(v), CTRL, 0xF, 0xF, true));
}
__device__ __forceinline__ float grp16_max(float v) {
  v = fmaxf(v, dpp_mov<0x161>(v));  // row_xmask:1
  v = fmaxf(v, dpp_mov<0x162>(v));  // row_xmask:2
  v = fmaxf(v, dpp_mov<0x164>(v));  // row_xmask:4
  v = fmaxf(v, dpp_mov<0x168>(v));  // row_xmask:8
  return v;
}
__device__ __forceinline__ float grp16_sum(float v) {
  v += dpp_mov<0x161>(v);
  v += dpp_mov<0x162>(v);
  v += dpp_mov<0x164>(v);
  v += dpp_mov<0x168>(v);
  return v;
}

// ---------------- Projection: out = in[R,128] @ W[128,WOUT], f16 output ----------
template <int WOUT, bool TR>
__global__ __launch_bounds__(128) void proj_kernel(const float* __restrict__ in,
                                                   const float* __restrict__ W,
                                                   _Float16* __restrict__ out,
                                                   float scale) {
  __shared__ __align__(16) _Float16 wt[WOUT][136];  // W^T, 16B-aligned row stride
  const int tid = threadIdx.x;
  for (int idx = tid; idx < 128 * WOUT; idx += 128) {
    int k = idx / WOUT, nn = idx - k * WOUT;
    wt[nn][k] = (_Float16)(W[idx] * scale);
  }
  __syncthreads();

  const int lane = tid & 31, wave = tid >> 5;
  const int g = lane >> 4, lh = lane & 15;
  const int r0 = (blockIdx.x * 4 + wave) * 16;

  const float* rowp = in + (size_t)(r0 + lh) * 128;
  v16h af[4];
#pragma unroll
  for (int kc = 0; kc < 4; ++kc) {
    V16U u;
    u.h[0] = cvt8(rowp + kc * 32 + g * 8);
    u.h[1] = cvt8(rowp + kc * 32 + 16 + g * 8);
    af[kc] = u.v;
  }

#pragma unroll
  for (int nt = 0; nt < WOUT / 16; ++nt) {
    v8f acc = {};
#pragma unroll
    for (int kc = 0; kc < 4; ++kc)
      acc = wmma_f16(af[kc], load_b32(&wt[nt * 16 + lh][kc * 32], g), acc);
#pragma unroll
    for (int j = 0; j < 8; ++j) {
      int row = r0 + j + 8 * g;
      int col = nt * 16 + lh;
      _Float16 hv = (_Float16)acc[j];
      if (TR) {
        int nb = row / VK_;
        int vk = row - nb * VK_;
        out[((size_t)nb * WOUT + col) * VK_ + vk] = hv;
      } else {
        out[(size_t)row * WOUT + col] = hv;
      }
    }
  }
}

// ---------------- Flash attention: 4 waves/block, 16 q-rows per wave --------------
// Double-buffered K/V tiles in LDS, one barrier per iteration, async copies of tile
// i+1 overlapped with WMMA compute on tile i, B-fragment loads software-pipelined.
__global__ __launch_bounds__(128) void attn_kernel(const _Float16* __restrict__ qh,
                                                   const _Float16* __restrict__ kh,
                                                   const _Float16* __restrict__ vT,
                                                   float* __restrict__ out) {
  __shared__ __align__(16) _Float16 ks[2][32][72];    // K tile: 32 kv rows x 64 h
  __shared__ __align__(16) _Float16 vs[2][C_][40];    // V^T tile: 128 c rows x 32 kv
  __shared__ __align__(16) _Float16 plds[4][16][40];  // per-wave P conversion buffer

  const int tid = threadIdx.x;
  const int lane = tid & 31, wave = tid >> 5;
  const int g = lane >> 4, lh = lane & 15;
  const int tile = blockIdx.x * 4 + wave;
  const int n  = tile / (VQ_ / 16);
  const int q0 = (tile % (VQ_ / 16)) * 16;

  const _Float16* qb = qh + ((size_t)n * VQ_ + q0) * H_;
  const _Float16* kb = kh + (size_t)n * VK_ * H_;
  const _Float16* vb = vT + (size_t)n * C_ * VK_;

  v16h qf[2];
  {
    const _Float16* rp = qb + lh * H_;
    qf[0] = load_a32(rp, g);
    qf[1] = load_a32(rp + 32, g);
  }

  v8f zf = {};
  v8f acc[8];
#pragma unroll
  for (int ct = 0; ct < 8; ++ct) acc[ct] = zf;
  float m[8], l[8];
#pragma unroll
  for (int j = 0; j < 8; ++j) { m[j] = -1.0e30f; l[j] = 0.0f; }

  _Float16(*pw)[40] = plds[wave];
  const int krow = tid >> 2, kpart = tid & 3;  // K tile: 32 rows x 4x16-half chunks
  const int ITERS = VK_ / 32;

#ifdef ASYNC_LDS
  {
    const _Float16* gk = kb + (size_t)krow * H_ + kpart * 16;
    _Float16* lk = &ks[0][krow][kpart * 16];
    async_cp16<0>(gk, lk);  async_cp16<16>(gk, lk);
    const _Float16* gv = vb + (size_t)tid * VK_;
    _Float16* lv = &vs[0][tid][0];
    async_cp16<0>(gv, lv);  async_cp16<16>(gv, lv);
    async_cp16<32>(gv, lv); async_cp16<48>(gv, lv);
  }
#else
  int4 tk0, tk1, tv0, tv1, tv2, tv3;
  {
    const int4* gk = (const int4*)(kb + (size_t)krow * H_ + kpart * 16);
    tk0 = gk[0]; tk1 = gk[1];
    const int4* gv = (const int4*)(vb + (size_t)tid * VK_);
    tv0 = gv[0]; tv1 = gv[1]; tv2 = gv[2]; tv3 = gv[3];
  }
  {
    int4* lk = (int4*)&ks[0][krow][kpart * 16];
    lk[0] = tk0; lk[1] = tk1;
    int4* lv = (int4*)&vs[0][tid][0];
    lv[0] = tv0; lv[1] = tv1; lv[2] = tv2; lv[3] = tv3;
  }
#endif

  for (int it = 0; it < ITERS; ++it) {
    const int cur = it & 1;
    const int kc0 = it * 32;
#ifdef ASYNC_LDS
    wait_async0();           // my copies for tile `it` landed
#endif
    __syncthreads();         // everyone's copies landed; prev reads of buf^1 done

    // ---- start fetching tile it+1 into the other buffer ----
    if (it + 1 < ITERS) {
      const int kn0 = kc0 + 32;
#ifdef ASYNC_LDS
      const _Float16* gk = kb + (size_t)(kn0 + krow) * H_ + kpart * 16;
      _Float16* lk = &ks[cur ^ 1][krow][kpart * 16];
      async_cp16<0>(gk, lk);  async_cp16<16>(gk, lk);
      const _Float16* gv = vb + (size_t)tid * VK_ + kn0;
      _Float16* lv = &vs[cur ^ 1][tid][0];
      async_cp16<0>(gv, lv);  async_cp16<16>(gv, lv);
      async_cp16<32>(gv, lv); async_cp16<48>(gv, lv);
#else
      const int4* gk = (const int4*)(kb + (size_t)(kn0 + krow) * H_ + kpart * 16);
      tk0 = gk[0]; tk1 = gk[1];
      const int4* gv = (const int4*)(vb + (size_t)tid * VK_ + kn0);
      tv0 = gv[0]; tv1 = gv[1]; tv2 = gv[2]; tv3 = gv[3];
#endif
    }

    // ---- S = Q * K^T : all four B fragments in distinct regs (loads overlap) ----
    const _Float16* kr0 = &ks[cur][lh][0];
    const _Float16* kr1 = &ks[cur][16 + lh][0];
    v16h bk0 = load_b32(kr0, g);
    v16h bk1 = load_b32(kr0 + 32, g);
    v16h bk2 = load_b32(kr1, g);
    v16h bk3 = load_b32(kr1 + 32, g);
    v8f s0 = wmma_f16(qf[0], bk0, zf);
    s0     = wmma_f16(qf[1], bk1, s0);
    v8f s1 = wmma_f16(qf[0], bk2, zf);
    s1     = wmma_f16(qf[1], bk3, s1);

    // ---- online softmax (log2 domain: q pre-scaled by H^-0.5 * log2(e)) ----
#pragma unroll
    for (int j = 0; j < 8; ++j) {
      float vj   = grp16_max(fmaxf(s0[j], s1[j]));
      float mnew = fmaxf(m[j], vj);
      float corr = exp2_fast(m[j] - mnew);
      float p0   = exp2_fast(s0[j] - mnew);
      float p1   = exp2_fast(s1[j] - mnew);
      m[j] = mnew;
      l[j] = l[j] * corr + grp16_sum(p0 + p1);
#pragma unroll
      for (int ct = 0; ct < 8; ++ct) acc[ct][j] *= corr;
      pw[j + 8 * g][lh]      = (_Float16)p0;  // D-layout -> row-major in LDS
      pw[j + 8 * g][16 + lh] = (_Float16)p1;
    }
    asm volatile("" ::: "memory");  // keep same-wave LDS stores before loads (DS in-order)
    v16h pf = load_a32(&pw[lh][0], g);

    // ---- O += P * V : software-pipelined B fragments ----
    v16h bv = load_b32(&vs[cur][lh][0], g);
#pragma unroll
    for (int ct = 0; ct < 8; ++ct) {
      v16h bnext;
      if (ct < 7) bnext = load_b32(&vs[cur][(ct + 1) * 16 + lh][0], g);
      acc[ct] = wmma_f16(pf, bv, acc[ct]);
      if (ct < 7) bv = bnext;
    }

#ifndef ASYNC_LDS
    if (it + 1 < ITERS) {  // park prefetched registers in the other buffer
      int4* lk = (int4*)&ks[cur ^ 1][krow][kpart * 16];
      lk[0] = tk0; lk[1] = tk1;
      int4* lv = (int4*)&vs[cur ^ 1][tid][0];
      lv[0] = tv0; lv[1] = tv1; lv[2] = tv2; lv[3] = tv3;
    }
#endif
  }

  float* ob = out + ((size_t)n * VQ_ + q0) * C_;
#pragma unroll
  for (int j = 0; j < 8; ++j) {
    float inv = 1.0f / l[j];
#pragma unroll
    for (int ct = 0; ct < 8; ++ct)
      ob[(size_t)(j + 8 * g) * C_ + ct * 16 + lh] = acc[ct][j] * inv;
  }
}

// ---------------- host launch ----------------
extern "C" void kernel_launch(void* const* d_in, const int* in_sizes, int n_in,
                              void* d_out, int out_size, void* d_ws, size_t ws_size,
                              hipStream_t stream) {
  const float* x  = (const float*)d_in[0];  // [N, VQ, C]
  const float* y  = (const float*)d_in[1];  // [N, VK, C]
  const float* Wq = (const float*)d_in[2];  // [C, H]
  const float* Wk = (const float*)d_in[3];  // [C, H]
  const float* Wv = (const float*)d_in[4];  // [C, C]
  float* out = (float*)d_out;               // [N, VQ, C]

  _Float16* qhp = (_Float16*)d_ws;                  // [N, VQ, H]  (2 MB)
  _Float16* khp = qhp + (size_t)N_ * VQ_ * H_;      // [N, VK, H]  (2 MB)
  _Float16* vtp = khp + (size_t)N_ * VK_ * H_;      // [N, C, VK]  (4 MB)

  const int R = N_ * VQ_;
  // H^-0.5 * log2(e): scores come out in log2 units, softmax uses raw v_exp_f32 (2^x)
  const float scale = 0.125f * 1.44269504088896340736f;

  proj_kernel<H_, false><<<R / 64, 128, 0, stream>>>(x, Wq, qhp, scale);
  proj_kernel<H_, false><<<R / 64, 128, 0, stream>>>(y, Wk, khp, 1.0f);
  proj_kernel<C_, true ><<<R / 64, 128, 0, stream>>>(y, Wv, vtp, 1.0f);
  attn_kernel<<<(N_ * VQ_ / 16) / 4, 128, 0, stream>>>(qhp, khp, vtp, out);
}
